// Time_Feature_20710332301988
// MI455X (gfx1250) — matrix operand order repaired and checked by reading, not compiled, not run on hardware
//
#include <hip/hip_runtime.h>
#include <math.h>

// ---------------- problem constants ----------------
#define GB   2            // batch
#define GH   32           // H
#define GW   64           // W
#define GM   4096         // B*H*W
#define PH   36           // H + 2*2 halo
#define PW   68           // W + 2*2 halo
#define HID  452
#define CIN  580
#define CPAD 608          // 580 padded to 19*32
#define KT5  19           // k-tiles per tap for GRU convs (608/32)
#define NZR  904          // z|r fused out channels
#define NZRP 960          // padded to 15*64
#define NQP  512          // 452 padded to 8*64
#define FINK 480          // 452 padded to 15*32
#define KTF  15
#define DIM3 256

typedef __attribute__((ext_vector_type(16))) _Float16 v16h;
typedef __attribute__((ext_vector_type(8)))  _Float16 v8h;
typedef __attribute__((ext_vector_type(8)))  float    v8f;

__device__ __forceinline__ v16h cat8(v8h lo, v8h hi) {
  return __builtin_shufflevector(lo, hi, 0,1,2,3,4,5,6,7,8,9,10,11,12,13,14,15);
}

// ---------------- generic im2col conv GEMM on WMMA f16 ----------------
// in : (GB, PH, PW, CPADT) f16 channels-last, zero halo + zero channel pad
// wt : [tap][ktile][n (NPAD cols)][32] f16, zero-padded
// out: (GM, OSTR) f32 (unpadded spatial), cols [0,NREAL) written
// MODE: 0 = 1x5 (dx = t-2), 1 = 5x1 (dy = t-2), 2 = 3x3 (dy=t/3-1, dx=t%3-1)
// Block: 128(M) x 64(N); 8 waves; each wave owns a 32x32 tile (4 accumulators).
template<int KTn, int TAPS, int MODE, int CPADT, int NPAD, int NREAL, int OSTR>
__global__ __launch_bounds__(256)
void conv_gemm(const _Float16* __restrict__ in, const _Float16* __restrict__ wt,
               float* __restrict__ out)
{
  const int lane = threadIdx.x & 31;
  const int wave = threadIdx.x >> 5;
  const int lhi  = lane >> 4;
  const int llo  = lane & 15;
  const int mbase = blockIdx.x * 128 + (wave & 3) * 32;
  const int nbase = blockIdx.y * 64 + (wave >> 2) * 32;

  // rows m0 = mbase+llo and m1 = m0+16 share the same image row y
  const int m0 = mbase + llo;
  const int b  = m0 >> 11;
  const int y  = (m0 >> 6) & (GH - 1);
  const int x0 = m0 & (GW - 1);

  const size_t pix0 = (size_t)(b * PH + (y + 2)) * PW + (x0 + 2);
  const _Float16* pa0 = in + pix0 * CPADT + (lhi ? 16 : 0);
  const _Float16* pa1 = pa0 + (size_t)16 * CPADT;
  const _Float16* pb  = wt + ((size_t)nbase + llo) * 32 + (lhi ? 16 : 0);

  v8f a00 = {}, a01 = {}, a10 = {}, a11 = {};

  for (int kt = 0; kt < KTn; ++kt) {
#pragma unroll
    for (int t = 0; t < TAPS; ++t) {
      const int dy = (MODE == 0) ? 0       : ((MODE == 1) ? (t - 2) : (t / 3 - 1));
      const int dx = (MODE == 0) ? (t - 2) : ((MODE == 1) ? 0       : (t % 3 - 1));
      const ptrdiff_t aoff = (ptrdiff_t)(dy * PW + dx) * CPADT;
      v16h A0 = cat8(*(const v8h*)(pa0 + aoff), *(const v8h*)(pa0 + aoff + 8));
      v16h A1 = cat8(*(const v8h*)(pa1 + aoff), *(const v8h*)(pa1 + aoff + 8));
      const _Float16* pbt = pb + (size_t)t * KTn * NPAD * 32;
      v16h B0 = *(const v16h*)(pbt);
      v16h B1 = *(const v16h*)(pbt + 16 * 32);
      a00 = __builtin_amdgcn_wmma_f32_16x16x32_f16(false, A0, false, B0, (short)0, a00, false, false);
      a01 = __builtin_amdgcn_wmma_f32_16x16x32_f16(false, A0, false, B1, (short)0, a01, false, false);
      a10 = __builtin_amdgcn_wmma_f32_16x16x32_f16(false, A1, false, B0, (short)0, a10, false, false);
      a11 = __builtin_amdgcn_wmma_f32_16x16x32_f16(false, A1, false, B1, (short)0, a11, false, false);
    }
    pa0 += 32;
    pa1 += 32;
    pb  += (size_t)NPAD * 32;
  }

  const int n0 = nbase + llo;
  const int n1 = n0 + 16;
  const int mr = mbase + (lhi << 3);
#pragma unroll
  for (int r = 0; r < 8; ++r) {
    float* po0 = out + (size_t)(mr + r) * OSTR;
    float* po1 = out + (size_t)(mr + 16 + r) * OSTR;
    if (n0 < NREAL) { po0[n0] = a00[r]; po1[n0] = a10[r]; }
    if (n1 < NREAL) { po0[n1] = a01[r]; po1[n1] = a11[r]; }
  }
}

// ---------------- all-pairs correlation GEMM (K=256) ----------------
// f1,f2: (B, 2048, 256) f16 channels-last; out: (B*2048, 2048) f32, scaled 1/16
__global__ __launch_bounds__(256)
void corr_gemm_wmma(const _Float16* __restrict__ f1, const _Float16* __restrict__ f2,
                    float* __restrict__ out)
{
  const int lane = threadIdx.x & 31;
  const int wave = threadIdx.x >> 5;
  const int lhi  = lane >> 4;
  const int llo  = lane & 15;
  const int bb   = blockIdx.z;
  const int mbase = blockIdx.x * 128 + (wave & 3) * 32;
  const int nbase = blockIdx.y * 64 + (wave >> 2) * 32;

  const _Float16* A  = f1 + (size_t)bb * 2048 * 256;
  const _Float16* Bp = f2 + (size_t)bb * 2048 * 256;

  v8f a00 = {}, a01 = {}, a10 = {}, a11 = {};
  const int aoff = lhi ? 16 : 0;
  const int boff = lhi ? 16 : 0;

  const _Float16* pa0 = A  + (size_t)(mbase + llo) * 256 + aoff;
  const _Float16* pa1 = pa0 + 16 * 256;
  const _Float16* pb0 = Bp + (size_t)(nbase + llo) * 256 + boff;
  const _Float16* pb1 = pb0 + 16 * 256;
#pragma unroll
  for (int kt = 0; kt < 8; ++kt) {
    v16h A0 = cat8(*(const v8h*)(pa0), *(const v8h*)(pa0 + 8));
    v16h A1 = cat8(*(const v8h*)(pa1), *(const v8h*)(pa1 + 8));
    v16h B0 = cat8(*(const v8h*)(pb0), *(const v8h*)(pb0 + 8));
    v16h B1 = cat8(*(const v8h*)(pb1), *(const v8h*)(pb1 + 8));
    a00 = __builtin_amdgcn_wmma_f32_16x16x32_f16(false, A0, false, B0, (short)0, a00, false, false);
    a01 = __builtin_amdgcn_wmma_f32_16x16x32_f16(false, A0, false, B1, (short)0, a01, false, false);
    a10 = __builtin_amdgcn_wmma_f32_16x16x32_f16(false, A1, false, B0, (short)0, a10, false, false);
    a11 = __builtin_amdgcn_wmma_f32_16x16x32_f16(false, A1, false, B1, (short)0, a11, false, false);
    pa0 += 32; pa1 += 32; pb0 += 32; pb1 += 32;
  }
  const int n0 = nbase + llo;
  const int mr = mbase + (lhi << 3);
#pragma unroll
  for (int r = 0; r < 8; ++r) {
    float* po0 = out + ((size_t)bb * 2048 + (mr + r)) * 2048;
    float* po1 = out + ((size_t)bb * 2048 + (mr + 16 + r)) * 2048;
    po0[n0]      = a00[r] * 0.0625f;   // 1/sqrt(256)
    po0[n0 + 16] = a01[r] * 0.0625f;
    po1[n0]      = a10[r] * 0.0625f;
    po1[n0 + 16] = a11[r] * 0.0625f;
  }
}

// ---------------- weight packing: OIHW f32 -> [tap][kt][n][32] f16 ----------------
__global__ void pack_weights(const float* __restrict__ src, _Float16* __restrict__ dst,
                             int O, int I, int T, int KTn, int Npadcols, int n0)
{
  size_t idx = (size_t)blockIdx.x * 256 + threadIdx.x;
  size_t total = (size_t)T * KTn * 32 * O;
  if (idx >= total) return;
  int o = (int)(idx % O);
  size_t r = idx / O;
  int i = (int)(r % (KTn * 32));
  int t = (int)(r / (KTn * 32));
  float v = (i < I) ? src[((size_t)o * I + i) * T + t] : 0.f;
  dst[(((size_t)t * KTn + (i >> 5)) * Npadcols + (n0 + o)) * 32 + (i & 31)] = (_Float16)v;
}

__global__ void zero_u32(unsigned int* p, size_t n) {
  size_t i = (size_t)blockIdx.x * 256 + threadIdx.x;
  if (i < n) p[i] = 0u;
}

// ---------------- feature prep ----------------
__global__ void prep(const float* __restrict__ x, _Float16* __restrict__ f1h,
                     _Float16* __restrict__ f2h, _Float16* __restrict__ inp16)
{
  size_t idx = (size_t)blockIdx.x * 256 + threadIdx.x;
  if (idx >= (size_t)GM * 256) return;
  int c = (int)(idx & 255);
  int m = (int)(idx >> 8);
  int b = m >> 11;
  int r = m & 2047;
  size_t base = ((size_t)(b * 2) * 2048 + r) * 256 + c;
  float v1 = x[base];
  float v2 = x[base + (size_t)2048 * 256];
  f1h[(size_t)m * 256 + c] = (_Float16)v1;
  f2h[(size_t)m * 256 + c] = (_Float16)v2;
  if (c >= 128) inp16[(size_t)m * 128 + (c - 128)] = (_Float16)fmaxf(v1, 0.f);
}

// ---------------- 2x2 average pool on (4096, h, w) maps ----------------
__global__ void avgpool2(const float* __restrict__ in, float* __restrict__ out, int hin, int win)
{
  int ho = hin >> 1, wo = win >> 1;
  size_t total = (size_t)GM * ho * wo;
  size_t idx = (size_t)blockIdx.x * 256 + threadIdx.x;
  if (idx >= total) return;
  int xo = (int)(idx % wo);
  size_t r = idx / wo;
  int yo = (int)(r % ho);
  int n  = (int)(r / ho);
  const float* p = in + ((size_t)n * hin + 2 * yo) * win + 2 * xo;
  out[idx] = 0.25f * (p[0] + p[1] + p[win] + p[win + 1]);
}

__device__ __forceinline__ float tapf(const float* img, int Hl, int Wl, float xi, float yi) {
  if (xi < 0.f || xi > (float)(Wl - 1) || yi < 0.f || yi > (float)(Hl - 1)) return 0.f;
  return img[(int)yi * Wl + (int)xi];
}

// ---------------- corr pyramid lookup + initial net ----------------
__global__ void lookup_init(const float* __restrict__ p0, const float* __restrict__ p1,
                            const float* __restrict__ p2, const float* __restrict__ p3,
                            const float* __restrict__ x, float* __restrict__ h)
{
  size_t idx = (size_t)blockIdx.x * 256 + threadIdx.x;
  if (idx >= (size_t)GM * HID) return;
  int c = (int)(idx % HID);
  int m = (int)(idx / HID);
  int b = m >> 11;
  int y = (m >> 6) & 31;
  int xq = m & 63;
  float val;
  if (c < 324) {
    int l = c / 81, rem = c % 81;
    int i = rem / 9, j = rem % 9;      // i -> x offset, j -> y offset (RAFT quirk)
    const float* img; int Hl, Wl;
    if (l == 0)      { img = p0; Hl = 32; Wl = 64; }
    else if (l == 1) { img = p1; Hl = 16; Wl = 32; }
    else if (l == 2) { img = p2; Hl = 8;  Wl = 16; }
    else             { img = p3; Hl = 4;  Wl = 8;  }
    img += (size_t)m * Hl * Wl;
    float sc = 1.f / (float)(1 << l);
    float px = (float)xq * sc + (float)(i - 4);
    float py = (float)y  * sc + (float)(j - 4);
    float x0 = floorf(px), y0 = floorf(py);
    float wx1 = px - x0, wx0 = 1.f - wx1;
    float wy1 = py - y0, wy0 = 1.f - wy1;
    val = tapf(img, Hl, Wl, x0,       y0      ) * (wx0 * wy0)
        + tapf(img, Hl, Wl, x0 + 1.f, y0      ) * (wx1 * wy0)
        + tapf(img, Hl, Wl, x0,       y0 + 1.f) * (wx0 * wy1)
        + tapf(img, Hl, Wl, x0 + 1.f, y0 + 1.f) * (wx1 * wy1);
  } else {
    val = tanhf(x[((size_t)(b * 2) * 2048 + (y * 64 + xq)) * 256 + (c - 324)]);
  }
  h[(size_t)m * HID + c] = val;
}

// ---------------- GRU elementwise (write into halo-padded f16 buffers) ----------------
__global__ void build_hx(const float* __restrict__ h, const _Float16* __restrict__ inp16,
                         _Float16* __restrict__ hx)
{
  size_t idx = (size_t)blockIdx.x * 256 + threadIdx.x;
  if (idx >= (size_t)GB * PH * PW * CPAD) return;
  int c = (int)(idx % CPAD);
  int p = (int)(idx / CPAD);
  int px = p % PW;
  int py = (p / PW) % PH;
  int b  = p / (PW * PH);
  int y = py - 2, x = px - 2;
  _Float16 v = (_Float16)0.f;
  if (y >= 0 && y < GH && x >= 0 && x < GW) {
    int m = b * 2048 + y * 64 + x;
    if (c < HID)      v = (_Float16)h[(size_t)m * HID + c];
    else if (c < CIN) v = inp16[(size_t)m * 128 + (c - HID)];
  }
  hx[idx] = v;
}

__global__ void gates_zr(const float* __restrict__ zr, const float* __restrict__ bz,
                         const float* __restrict__ br, const float* __restrict__ h,
                         const _Float16* __restrict__ inp16,
                         float* __restrict__ zbuf, _Float16* __restrict__ qin)
{
  size_t idx = (size_t)blockIdx.x * 256 + threadIdx.x;
  if (idx >= (size_t)GB * PH * PW * CPAD) return;
  int c = (int)(idx % CPAD);
  int p = (int)(idx / CPAD);
  int px = p % PW;
  int py = (p / PW) % PH;
  int b  = p / (PW * PH);
  int y = py - 2, x = px - 2;
  if (y < 0 || y >= GH || x < 0 || x >= GW) { qin[idx] = (_Float16)0.f; return; }
  int m = b * 2048 + y * 64 + x;
  if (c < HID) {
    float zp = zr[(size_t)m * NZR + c] + bz[c];
    float rp = zr[(size_t)m * NZR + HID + c] + br[c];
    float z = 1.f / (1.f + expf(-zp));
    float r = 1.f / (1.f + expf(-rp));
    zbuf[(size_t)m * HID + c] = z;
    qin[idx] = (_Float16)(r * h[(size_t)m * HID + c]);
  } else if (c < CIN) {
    qin[idx] = inp16[(size_t)m * 128 + (c - HID)];
  } else {
    qin[idx] = (_Float16)0.f;
  }
}

__global__ void update_h(const float* __restrict__ qpre, const float* __restrict__ bq,
                         const float* __restrict__ zbuf, float* __restrict__ h)
{
  size_t idx = (size_t)blockIdx.x * 256 + threadIdx.x;
  if (idx >= (size_t)GM * HID) return;
  int c = (int)(idx % HID);
  float q = tanhf(qpre[idx] + bq[c]);
  float z = zbuf[idx];
  float hv = h[idx];
  h[idx] = (1.f - z) * hv + z * q;
}

__global__ void build_fin(const float* __restrict__ h, _Float16* __restrict__ fin)
{
  size_t idx = (size_t)blockIdx.x * 256 + threadIdx.x;
  if (idx >= (size_t)GB * PH * PW * FINK) return;
  int c = (int)(idx % FINK);
  int p = (int)(idx / FINK);
  int px = p % PW;
  int py = (p / PW) % PH;
  int b  = p / (PW * PH);
  int y = py - 2, x = px - 2;
  _Float16 v = (_Float16)0.f;
  if (y >= 0 && y < GH && x >= 0 && x < GW && c < HID) {
    v = (_Float16)h[((size_t)(b * 2048 + y * 64 + x)) * HID + c];
  }
  fin[idx] = v;
}

__global__ void final_out(const float* __restrict__ pre, const float* __restrict__ bc,
                          float* __restrict__ out)
{
  size_t idx = (size_t)blockIdx.x * 256 + threadIdx.x;
  if (idx >= (size_t)GB * DIM3 * 16 * 32) return;
  int x2 = (int)(idx & 31);
  int y2 = (int)((idx >> 5) & 15);
  int c  = (int)((idx >> 9) & 255);
  int b  = (int)(idx >> 17);
  float bcv = bc[c];
  float s = 0.f;
  for (int dy = 0; dy < 2; ++dy)
    for (int dx = 0; dx < 2; ++dx) {
      int m = b * 2048 + (2 * y2 + dy) * 64 + (2 * x2 + dx);
      s += fmaxf(pre[(size_t)m * DIM3 + c] + bcv, 0.f);
    }
  out[idx] = 0.25f * s;
}

// ---------------- host orchestration ----------------
static inline int nblk(size_t total) { return (int)((total + 255) / 256); }

extern "C" void kernel_launch(void* const* d_in, const int* in_sizes, int n_in,
                              void* d_out, int out_size, void* d_ws, size_t ws_size,
                              hipStream_t stream) {
  const float* x   = (const float*)d_in[0];
  const float* wz1 = (const float*)d_in[3];  const float* bz1 = (const float*)d_in[4];
  const float* wr1 = (const float*)d_in[5];  const float* br1 = (const float*)d_in[6];
  const float* wq1 = (const float*)d_in[7];  const float* bq1 = (const float*)d_in[8];
  const float* wz2 = (const float*)d_in[9];  const float* bz2 = (const float*)d_in[10];
  const float* wr2 = (const float*)d_in[11]; const float* br2 = (const float*)d_in[12];
  const float* wq2 = (const float*)d_in[13]; const float* bq2 = (const float*)d_in[14];
  const float* wc  = (const float*)d_in[15]; const float* bc  = (const float*)d_in[16];
  float* out = (float*)d_out;

  char* ws = (char*)d_ws;
  size_t off = 0;
  auto carve = [&](size_t bytes) -> void* {
    off = (off + 255) & ~(size_t)255;
    void* p = ws + off;
    off += bytes;
    return p;
  };

  _Float16* f1h   = (_Float16*)carve((size_t)GM * 256 * 2);
  _Float16* f2h   = (_Float16*)carve((size_t)GM * 256 * 2);
  _Float16* inp16 = (_Float16*)carve((size_t)GM * 128 * 2);
  float* pyr0 = (float*)carve((size_t)GM * 2048 * 4);
  float* pyr1 = (float*)carve((size_t)GM * 16 * 32 * 4);
  float* pyr2 = (float*)carve((size_t)GM * 8 * 16 * 4);
  float* pyr3 = (float*)carve((size_t)GM * 4 * 8 * 4);
  float* hbuf = (float*)carve((size_t)GM * HID * 4);
  _Float16* hx16  = (_Float16*)carve((size_t)GB * PH * PW * CPAD * 2);
  _Float16* qin16 = (_Float16*)carve((size_t)GB * PH * PW * CPAD * 2);
  float* zrbuf = (float*)carve((size_t)GM * NZR * 4);
  float* zbuf  = (float*)carve((size_t)GM * HID * 4);
  float* qbuf  = (float*)carve((size_t)GM * HID * 4);
  _Float16* fin16 = (_Float16*)carve((size_t)GB * PH * PW * FINK * 2);
  float* finpre   = (float*)carve((size_t)GM * DIM3 * 4);

  const size_t szZR = (size_t)5 * KT5 * NZRP * 32;  // halves
  const size_t szQ  = (size_t)5 * KT5 * NQP * 32;
  const size_t szF  = (size_t)9 * KTF * DIM3 * 32;
  _Float16* Wzr15 = (_Float16*)carve(szZR * 2);
  _Float16* Wq15  = (_Float16*)carve(szQ * 2);
  _Float16* Wzr51 = (_Float16*)carve(szZR * 2);
  _Float16* Wq51  = (_Float16*)carve(szQ * 2);
  _Float16* Wfin  = (_Float16*)carve(szF * 2);

  // zero the (contiguous) packed-weight region (padding columns/channels must be 0)
  size_t wu32 = (szZR * 2 + szQ * 2 + szF) * 2 / 4;
  zero_u32<<<nblk(wu32), 256, 0, stream>>>((unsigned int*)Wzr15, wu32);

  // pack weights: OIHW flattens to (o*I + i)*T + t for 1x5, 5x1 and 3x3 alike
  pack_weights<<<nblk((size_t)5 * KT5 * 32 * HID), 256, 0, stream>>>(wz1, Wzr15, HID, CIN, 5, KT5, NZRP, 0);
  pack_weights<<<nblk((size_t)5 * KT5 * 32 * HID), 256, 0, stream>>>(wr1, Wzr15, HID, CIN, 5, KT5, NZRP, HID);
  pack_weights<<<nblk((size_t)5 * KT5 * 32 * HID), 256, 0, stream>>>(wq1, Wq15,  HID, CIN, 5, KT5, NQP, 0);
  pack_weights<<<nblk((size_t)5 * KT5 * 32 * HID), 256, 0, stream>>>(wz2, Wzr51, HID, CIN, 5, KT5, NZRP, 0);
  pack_weights<<<nblk((size_t)5 * KT5 * 32 * HID), 256, 0, stream>>>(wr2, Wzr51, HID, CIN, 5, KT5, NZRP, HID);
  pack_weights<<<nblk((size_t)5 * KT5 * 32 * HID), 256, 0, stream>>>(wq2, Wq51,  HID, CIN, 5, KT5, NQP, 0);
  pack_weights<<<nblk((size_t)9 * KTF * 32 * DIM3), 256, 0, stream>>>(wc, Wfin, DIM3, HID, 9, KTF, DIM3, 0);

  // features
  prep<<<nblk((size_t)GM * 256), 256, 0, stream>>>(x, f1h, f2h, inp16);

  // all-pairs correlation -> pyramid
  corr_gemm_wmma<<<dim3(16, 32, GB), 256, 0, stream>>>(f1h, f2h, pyr0);
  avgpool2<<<nblk((size_t)GM * 16 * 32), 256, 0, stream>>>(pyr0, pyr1, 32, 64);
  avgpool2<<<nblk((size_t)GM * 8 * 16), 256, 0, stream>>>(pyr1, pyr2, 16, 32);
  avgpool2<<<nblk((size_t)GM * 4 * 8), 256, 0, stream>>>(pyr2, pyr3, 8, 16);

  // corr lookup (coords = identity grid) + tanh part -> initial hidden state
  lookup_init<<<nblk((size_t)GM * HID), 256, 0, stream>>>(pyr0, pyr1, pyr2, pyr3, x, hbuf);

  const size_t nPadC = (size_t)GB * PH * PW * CPAD;

  // 12 GRU iterations (SepConvGRU: 1x5 half then 5x1 half)
  for (int it = 0; it < 12; ++it) {
    // ---- half 0: 1x5 taps ----
    build_hx<<<nblk(nPadC), 256, 0, stream>>>(hbuf, inp16, hx16);
    conv_gemm<KT5, 5, 0, CPAD, NZRP, NZR, NZR>
        <<<dim3(GM / 128, NZRP / 64), 256, 0, stream>>>(hx16, Wzr15, zrbuf);
    gates_zr<<<nblk(nPadC), 256, 0, stream>>>(zrbuf, bz1, br1, hbuf, inp16, zbuf, qin16);
    conv_gemm<KT5, 5, 0, CPAD, NQP, HID, HID>
        <<<dim3(GM / 128, NQP / 64), 256, 0, stream>>>(qin16, Wq15, qbuf);
    update_h<<<nblk((size_t)GM * HID), 256, 0, stream>>>(qbuf, bq1, zbuf, hbuf);

    // ---- half 1: 5x1 taps ----
    build_hx<<<nblk(nPadC), 256, 0, stream>>>(hbuf, inp16, hx16);
    conv_gemm<KT5, 5, 1, CPAD, NZRP, NZR, NZR>
        <<<dim3(GM / 128, NZRP / 64), 256, 0, stream>>>(hx16, Wzr51, zrbuf);
    gates_zr<<<nblk(nPadC), 256, 0, stream>>>(zrbuf, bz2, br2, hbuf, inp16, zbuf, qin16);
    conv_gemm<KT5, 5, 1, CPAD, NQP, HID, HID>
        <<<dim3(GM / 128, NQP / 64), 256, 0, stream>>>(qin16, Wq51, qbuf);
    update_h<<<nblk((size_t)GM * HID), 256, 0, stream>>>(qbuf, bq2, zbuf, hbuf);
  }

  // final 3x3 conv + relu + avgpool 2x2
  build_fin<<<nblk((size_t)GB * PH * PW * FINK), 256, 0, stream>>>(hbuf, fin16);
  conv_gemm<KTF, 9, 2, FINK, DIM3, DIM3, DIM3>
      <<<dim3(GM / 128, DIM3 / 64), 256, 0, stream>>>(fin16, Wfin, finpre);
  final_out<<<nblk((size_t)GB * DIM3 * 16 * 32), 256, 0, stream>>>(finpre, bc, out);
}
